// Qwen2MoeExpert_53953379173325
// MI455X (gfx1250) — compile-verified, matrix operand order checked by reading, not run on hardware
//
#include <hip/hip_runtime.h>

typedef __attribute__((ext_vector_type(2)))  _Float16 v2h;
typedef __attribute__((ext_vector_type(8)))  _Float16 v8h;
typedef __attribute__((ext_vector_type(16))) _Float16 v16h;
typedef __attribute__((ext_vector_type(8)))  float    v8f;

#define TOKENS  8192
#define HIDDEN  2048
#define INTER   5632
#define LDSK    40          // padded K-stride (fp16 elems) for a 32-deep K tile
#define TILE    (64 * LDSK) // one 64-col x 32-K weight tile in LDS

// Assemble a 16x32 fp16 WMMA fragment from two 16B chunks:
// halves 0..7 at p[0..7], halves 8..15 at p[16..23] (ISA 16-bit A/B layout).
__device__ __forceinline__ v16h load_frag16(const _Float16* p) {
    v8h lo = *(const v8h*)(p);
    v8h hi = *(const v8h*)(p + 16);
    v16h r;
#pragma unroll
    for (int i = 0; i < 8; ++i) { r[i] = lo[i]; r[i + 8] = hi[i]; }
    return r;
}

// Packed GPTQ int4 dequant via fp16 magic numbers.
// (nib|0x6400) == fp16(1024+q) exactly; (1024+q)-(1024+z1) is EXACT in fp16,
// then one packed mul by s.  K pairs land in slot order (0,4),(1,5),(2,6),(3,7);
// the A side is stored with the same permutation so the dot product is unchanged.
__device__ __forceinline__ void dequant8_store(unsigned qw, v2h spk, v2h zpk,
                                               _Float16* dst) {
    const unsigned MASK = 0x000F000Fu, MAGIC = 0x64006400u;
    v2h w0 = (__builtin_bit_cast(v2h, ( qw         & MASK) | MAGIC) - zpk) * spk;
    v2h w1 = (__builtin_bit_cast(v2h, ((qw >> 4)   & MASK) | MAGIC) - zpk) * spk;
    v2h w2 = (__builtin_bit_cast(v2h, ((qw >> 8)   & MASK) | MAGIC) - zpk) * spk;
    v2h w3 = (__builtin_bit_cast(v2h, ((qw >> 12)  & MASK) | MAGIC) - zpk) * spk;
    v8h o;
    o[0] = w0[0]; o[1] = w0[1]; o[2] = w1[0]; o[3] = w1[1];
    o[4] = w2[0]; o[5] = w2[1]; o[6] = w3[0]; o[7] = w3[1];
    *(v8h*)dst = o;
}

__device__ __forceinline__ v2h make_zpk(int zpacked, int n) {
    int z1 = ((zpacked >> ((n & 7) * 4)) & 0xF) + 1;
    _Float16 zm = __builtin_bit_cast(_Float16, (unsigned short)(0x6400u + z1));
    v2h r = {zm, zm};
    return r;
}

__device__ __forceinline__ v2h splat_h(float s) {
    _Float16 h = (_Float16)s;
    v2h r = {h, h};
    return r;
}

// x fp32 -> fp16, K pairs interleaved (0,4),(1,5),(2,6),(3,7) per 8-group to
// match the packed dequant slot order.
__global__ __launch_bounds__(256)
void cvt_kernel(const float* __restrict__ x, _Float16* __restrict__ xb) {
    int idx = blockIdx.x * 256 + threadIdx.x;
    const float4* x4 = (const float4*)x;
    float4 a = x4[idx * 2];       // k = 0..3 of this 8-group
    float4 b = x4[idx * 2 + 1];   // k = 4..7
    v8h o;
    o[0] = (_Float16)a.x; o[1] = (_Float16)b.x;
    o[2] = (_Float16)a.y; o[3] = (_Float16)b.y;
    o[4] = (_Float16)a.z; o[5] = (_Float16)b.z;
    o[6] = (_Float16)a.w; o[7] = (_Float16)b.w;
    ((v8h*)xb)[idx] = o;
}

// Fused gate+up GEMM, double-buffered LDS, software-pipelined dequant.
// Block = 8 waves, computes 128 (tokens) x 64 (inter) of BOTH projections.
__global__ __launch_bounds__(256)
void gateup_kernel(const _Float16* __restrict__ xb,
                   const unsigned* __restrict__ gq, const int* __restrict__ gz,
                   const float* __restrict__ gs,
                   const unsigned* __restrict__ uq, const int* __restrict__ uz,
                   const float* __restrict__ us,
                   _Float16* __restrict__ hb) {
    __shared__ _Float16 ldsG[2][TILE];
    __shared__ _Float16 ldsU[2][TILE];

    const int t    = threadIdx.x;
    const int wave = t >> 5;
    const int lane = t & 31;
    const int m0   = blockIdx.y * 128 + wave * 16;
    const int n0   = blockIdx.x * 64;

    // dequant mapping: 256 threads cover 64 cols x 4 packed rows (= 32 K)
    const int nl   = t & 63;
    const int pr   = t >> 6;
    const int n    = n0 + nl;
    const int woff = nl * LDSK + pr * 8;

    // WMMA fragment mapping (wave32, 16-bit operands)
    const int frow  = lane & 15;
    const int khalf = (lane >> 4) * 8;
    const _Float16* xrow = xb + (m0 + frow) * HIDDEN + khalf;

    v8f accg[4] = {};
    v8f accu[4] = {};

    v2h sgpk, zgpk, supk, zupk;
    auto loadc = [&](int g) {
        sgpk = splat_h(gs[g * INTER + n]);
        supk = splat_h(us[g * INTER + n]);
        zgpk = make_zpk(gz[g * (INTER / 8) + (n >> 3)], n);
        zupk = make_zpk(uz[g * (INTER / 8) + (n >> 3)], n);
    };

    // prologue: constants for group 0, dequant K-step 0 into buffer 0
    loadc(0);
    dequant8_store(gq[pr * INTER + n], sgpk, zgpk, &ldsG[0][woff]);
    dequant8_store(uq[pr * INTER + n], supk, zupk, &ldsU[0][woff]);

    const int NSTEP = HIDDEN / 32;   // 64
#pragma unroll 4
    for (int ks = 0; ks < NSTEP; ++ks) {
        __syncthreads();             // buffer ks&1 ready; reads of (ks-1)&1 done
        const int nxt = ks + 1;
        unsigned nqg = 0, nqu = 0;
        if (nxt < NSTEP) {           // issue next step's weight loads early
            const int kp = nxt * 4 + pr;
            nqg = gq[kp * INTER + n];
            nqu = uq[kp * INTER + n];
        }

        // Hoist A + ALL B fragments into distinct registers so every
        // ds_load_b128 issues up front and the 8 WMMAs run back-to-back.
        const _Float16* bG = &ldsG[ks & 1][frow * LDSK + khalf];
        const _Float16* bU = &ldsU[ks & 1][frow * LDSK + khalf];
        v16h a = load_frag16(xrow + ks * 32);
        v16h bgf[4], buf[4];
#pragma unroll
        for (int nt = 0; nt < 4; ++nt) {
            bgf[nt] = load_frag16(bG + nt * 16 * LDSK);
            buf[nt] = load_frag16(bU + nt * 16 * LDSK);
        }
#pragma unroll
        for (int nt = 0; nt < 4; ++nt) {
            accg[nt] = __builtin_amdgcn_wmma_f32_16x16x32_f16(
                false, a, false, bgf[nt], (short)0, accg[nt], false, false);
            accu[nt] = __builtin_amdgcn_wmma_f32_16x16x32_f16(
                false, a, false, buf[nt], (short)0, accu[nt], false, false);
        }

        if (nxt < NSTEP) {
            if ((nxt & 3) == 0) loadc(nxt >> 2);   // new group of 128 K
            dequant8_store(nqg, sgpk, zgpk, &ldsG[nxt & 1][woff]);
            dequant8_store(nqu, supk, zupk, &ldsU[nxt & 1][woff]);
        }
    }

    // Epilogue: h = silu(g)*u, stored fp16 with the K-pair permutation applied
    // along I (h is the A/K side of the down GEMM).
    const int l15   = lane & 15;
    const int j     = l15 & 7;
    const int permc = (l15 & 8) + (((j & 3) << 1) | (j >> 2));
    const int obase = m0 + (lane >> 4) * 8;
#pragma unroll
    for (int nt = 0; nt < 4; ++nt) {
#pragma unroll
        for (int r = 0; r < 8; ++r) {
            float gv = accg[nt][r];
            float uv = accu[nt][r];
            float hv = (gv / (1.0f + __expf(-gv))) * uv;
            hb[(obase + r) * INTER + n0 + nt * 16 + permc] = (_Float16)hv;
        }
    }
}

// Down projection: [T, I] fp16 (K-permuted) @ dequant(int4 [I, H]) -> fp32 out
__global__ __launch_bounds__(256)
void down_kernel(const _Float16* __restrict__ hb,
                 const unsigned* __restrict__ dq, const int* __restrict__ dz,
                 const float* __restrict__ ds,
                 float* __restrict__ out) {
    __shared__ _Float16 ldsW[2][TILE];

    const int t    = threadIdx.x;
    const int wave = t >> 5;
    const int lane = t & 31;
    const int m0   = blockIdx.y * 128 + wave * 16;
    const int n0   = blockIdx.x * 64;

    const int nl   = t & 63;
    const int pr   = t >> 6;
    const int n    = n0 + nl;
    const int woff = nl * LDSK + pr * 8;

    const int frow  = lane & 15;
    const int khalf = (lane >> 4) * 8;
    const _Float16* hrow = hb + (m0 + frow) * INTER + khalf;

    v8f acc[4] = {};

    v2h spk, zpk;
    auto loadc = [&](int g) {
        spk = splat_h(ds[g * HIDDEN + n]);
        zpk = make_zpk(dz[g * (HIDDEN / 8) + (n >> 3)], n);
    };

    loadc(0);
    dequant8_store(dq[pr * HIDDEN + n], spk, zpk, &ldsW[0][woff]);

    const int NSTEP = INTER / 32;    // 176
#pragma unroll 4
    for (int ks = 0; ks < NSTEP; ++ks) {
        __syncthreads();
        const int nxt = ks + 1;
        unsigned nq = 0;
        if (nxt < NSTEP) {
            nq = dq[(nxt * 4 + pr) * HIDDEN + n];
        }

        const _Float16* bW = &ldsW[ks & 1][frow * LDSK + khalf];
        v16h a = load_frag16(hrow + ks * 32);
        v16h bf[4];
#pragma unroll
        for (int nt = 0; nt < 4; ++nt) {
            bf[nt] = load_frag16(bW + nt * 16 * LDSK);
        }
#pragma unroll
        for (int nt = 0; nt < 4; ++nt) {
            acc[nt] = __builtin_amdgcn_wmma_f32_16x16x32_f16(
                false, a, false, bf[nt], (short)0, acc[nt], false, false);
        }

        if (nxt < NSTEP) {
            if ((nxt & 3) == 0) loadc(nxt >> 2);
            dequant8_store(nq, spk, zpk, &ldsW[nxt & 1][woff]);
        }
    }

    const int ocol  = n0 + (lane & 15);
    const int obase = m0 + (lane >> 4) * 8;
#pragma unroll
    for (int nt = 0; nt < 4; ++nt) {
#pragma unroll
        for (int r = 0; r < 8; ++r) {
            out[(obase + r) * HIDDEN + ocol + nt * 16] = acc[nt][r];
        }
    }
}

extern "C" void kernel_launch(void* const* d_in, const int* in_sizes, int n_in,
                              void* d_out, int out_size, void* d_ws, size_t ws_size,
                              hipStream_t stream) {
    (void)in_sizes; (void)n_in; (void)out_size; (void)ws_size;
    const float*    x  = (const float*)d_in[0];
    const unsigned* gq = (const unsigned*)d_in[1];
    const int*      gz = (const int*)d_in[2];
    const float*    gs = (const float*)d_in[3];
    const unsigned* uq = (const unsigned*)d_in[4];
    const int*      uz = (const int*)d_in[5];
    const float*    us = (const float*)d_in[6];
    const unsigned* dq = (const unsigned*)d_in[7];
    const int*      dz = (const int*)d_in[8];
    const float*    ds = (const float*)d_in[9];
    float* out = (float*)d_out;

    _Float16* xb = (_Float16*)d_ws;                                          // 33.5 MB
    _Float16* hb = (_Float16*)((char*)d_ws + (size_t)TOKENS * HIDDEN * 2);   // 92.3 MB

    cvt_kernel<<<(TOKENS * HIDDEN) / (256 * 8), 256, 0, stream>>>(x, xb);
    gateup_kernel<<<dim3(INTER / 64, TOKENS / 128), 256, 0, stream>>>(
        xb, gq, gz, gs, uq, uz, us, hb);
    down_kernel<<<dim3(HIDDEN / 64, TOKENS / 128), 256, 0, stream>>>(
        hb, dq, dz, ds, out);
}
